// Merge_45732811767879
// MI455X (gfx1250) — matrix operand order repaired and checked by reading, not compiled, not run on hardware
//
#include <hip/hip_runtime.h>

typedef __attribute__((ext_vector_type(16))) __bf16 v16bf;
typedef __attribute__((ext_vector_type(8)))  float  v8f;

#define B_DIM    512
#define N_DIM    1024
#define C_DIM    192
#define KEEP     256
#define COMPRESS 512   // == B in the reference (unimp.shape[0])
#define NSPLIT   8

// ---------------- Kernel 1a: partial sum of squares over token axis ----------------
__global__ void __launch_bounds__(C_DIM)
k_partial_sumsq(const float* __restrict__ x, float* __restrict__ part) {
    int b = blockIdx.x;
    int s = blockIdx.y;          // n-split 0..NSPLIT-1
    int c = threadIdx.x;         // 0..191
    const float* xp = x + ((size_t)b * N_DIM + (size_t)s * (N_DIM / NSPLIT)) * C_DIM + c;
    float acc = 0.f;
    for (int n = 0; n < N_DIM / NSPLIT; ++n) {
        float v = xp[(size_t)n * C_DIM];
        acc += v * v;
    }
    part[((size_t)b * NSPLIT + s) * C_DIM + c] = acc;
}

// ---------------- Kernel 1b: reduce partials -> 1/||x[b,:,c]|| ----------------
__global__ void __launch_bounds__(256)
k_invnorm(const float* __restrict__ part, float* __restrict__ invn) {
    int i = blockIdx.x * blockDim.x + threadIdx.x;  // b*C_DIM + c
    int b = i / C_DIM, c = i % C_DIM;
    float s = 0.f;
#pragma unroll
    for (int k = 0; k < NSPLIT; ++k) s += part[((size_t)b * NSPLIT + k) * C_DIM + c];
    invn[i] = rsqrtf(s);
}

// ---------------- Kernel 2: batched sim GEMM (bf16 WMMA) + row argmax ----------------
// A element = x[u,c] * invn[c]^2  (normalization folded to one side; argmax-invariant)
// B element = x[k,c]              (raw, bf16)
__device__ __forceinline__ void load8_scaled(const float* __restrict__ xp,
                                             const float* __restrict__ ip,
                                             v16bf& d, int e0) {
    const float4* x4 = (const float4*)xp;
    const float4* i4 = (const float4*)ip;
    float4 a = x4[0], b = x4[1], s = i4[0], t = i4[1];
    d[e0 + 0] = (__bf16)(a.x * s.x * s.x);
    d[e0 + 1] = (__bf16)(a.y * s.y * s.y);
    d[e0 + 2] = (__bf16)(a.z * s.z * s.z);
    d[e0 + 3] = (__bf16)(a.w * s.w * s.w);
    d[e0 + 4] = (__bf16)(b.x * t.x * t.x);
    d[e0 + 5] = (__bf16)(b.y * t.y * t.y);
    d[e0 + 6] = (__bf16)(b.z * t.z * t.z);
    d[e0 + 7] = (__bf16)(b.w * t.w * t.w);
}

__device__ __forceinline__ void load8_raw(const float* __restrict__ xp, v16bf& d, int e0) {
    const float4* x4 = (const float4*)xp;
    float4 a = x4[0], b = x4[1];
    d[e0 + 0] = (__bf16)a.x;  d[e0 + 1] = (__bf16)a.y;
    d[e0 + 2] = (__bf16)a.z;  d[e0 + 3] = (__bf16)a.w;
    d[e0 + 4] = (__bf16)b.x;  d[e0 + 5] = (__bf16)b.y;
    d[e0 + 6] = (__bf16)b.z;  d[e0 + 7] = (__bf16)b.w;
}

__global__ void __launch_bounds__(256)
k_sim_argmax(const float* __restrict__ x, const float* __restrict__ invn,
             int* __restrict__ dstIdx) {
    int b      = blockIdx.y;
    int grp    = blockIdx.x;            // 0..3
    int wave   = threadIdx.x >> 5;      // 0..7  (wave32)
    int lane   = threadIdx.x & 31;
    int lane16 = lane & 15;
    bool hi    = lane >= 16;
    int utile  = grp * 8 + wave;        // 0..31  (512 u-rows / 16)

    const float* xb  = x + (size_t)b * N_DIM * C_DIM;
    const float* ivp = invn + (size_t)b * C_DIM;

    // A fragment (16x32 bf16 per K-chunk): lane holds row M=lane16,
    // K halves split by lane<16 / >=16: contiguous k = 16h + s, same order as B.
    const float* up = xb + (size_t)(KEEP + utile * 16 + lane16) * C_DIM;
    v16bf afrag[6];
#pragma unroll
    for (int q = 0; q < 6; ++q) {
        int c0 = q * 32 + (hi ? 16 : 0);
        int c1 = q * 32 + (hi ? 24 : 8);
        load8_scaled(up + c0, ivp + c0, afrag[q], 0);
        load8_scaled(up + c1, ivp + c1, afrag[q], 8);
    }

    float bestV[8];
    int   bestI[8];
#pragma unroll
    for (int e = 0; e < 8; ++e) { bestV[e] = -__builtin_inff(); bestI[e] = 0; }

    for (int kt = 0; kt < 16; ++kt) {
        // B fragment (32x16 bf16 per K-chunk): lane = column N = token kt*16+lane16,
        // K = (hi?16:0)+j within the 32-chunk → 16 contiguous channels per lane.
        const float* kp = xb + (size_t)(kt * 16 + lane16) * C_DIM;
        v8f acc = {};
#pragma unroll
        for (int q = 0; q < 6; ++q) {
            v16bf bfr;
            const float* p = kp + q * 32 + (hi ? 16 : 0);
            load8_raw(p,     bfr, 0);
            load8_raw(p + 8, bfr, 8);
            acc = __builtin_amdgcn_wmma_f32_16x16x32_bf16(
                false, afrag[q], false, bfr, (short)0, acc, false, false);
        }
        int kcol = kt * 16 + lane16;
#pragma unroll
        for (int e = 0; e < 8; ++e) {
            float v = acc[e];
            if (kcol == 0) v = -__builtin_inff();   // sim[:, :, 0] = -inf
            if (v > bestV[e]) { bestV[e] = v; bestI[e] = kcol; }
        }
    }

    // Reduce over the 16 lanes holding different k columns (stays within half-wave).
#pragma unroll
    for (int e = 0; e < 8; ++e) {
        float v = bestV[e]; int i = bestI[e];
#pragma unroll
        for (int m = 1; m < 16; m <<= 1) {
            float ov = __shfl_xor(v, m, 32);
            int   oi = __shfl_xor(i, m, 32);
            if (ov > v || (ov == v && oi < i)) { v = ov; i = oi; }  // jnp.argmax tie-break
        }
        bestV[e] = v; bestI[e] = i;
    }
    if (lane16 == 0) {  // lane 0 -> rows M=0..7, lane 16 -> rows M=8..15
        int rowBase = utile * 16 + (hi ? 8 : 0);
#pragma unroll
        for (int e = 0; e < 8; ++e)
            dstIdx[b * COMPRESS + rowBase + e] = bestI[e];
    }
}

// ---------------- Kernel 3: deterministic scatter-merge ----------------
__global__ void __launch_bounds__(256)
k_merge(const float* __restrict__ x, const int* __restrict__ dstIdx,
        float* __restrict__ out) {
    __shared__ int s_idx[COMPRESS];
    __shared__ int s_cnt[KEEP];
    __shared__ int s_off[KEEP];
    __shared__ int s_list[COMPRESS];
    int b = blockIdx.x;
    int t = threadIdx.x;

    s_idx[t]       = dstIdx[b * COMPRESS + t];
    s_idx[t + 256] = dstIdx[b * COMPRESS + t + 256];
    s_cnt[t] = 0;
    __syncthreads();
    atomicAdd(&s_cnt[s_idx[t]], 1);          // integer LDS atomics: deterministic counts
    atomicAdd(&s_cnt[s_idx[t + 256]], 1);
    __syncthreads();
    if (t == 0) {
        int run = 0;
        for (int k = 0; k < KEEP; ++k) { s_off[k] = run; run += s_cnt[k]; }
    }
    __syncthreads();
    {   // bucket fill: thread t owns bucket k=t, writes its own region in ascending u
        int cur = s_off[t];
        for (int u = 0; u < COMPRESS; ++u)
            if (s_idx[u] == t) s_list[cur++] = u;
    }
    __syncthreads();

    const float* xb = x   + (size_t)b * N_DIM * C_DIM;
    float*       ob = out + (size_t)b * KEEP * C_DIM;
    if (t < C_DIM) {
        for (int k = 0; k < KEEP; ++k) {
            int cnt = s_cnt[k];
            int off = s_off[k];
            float acc = xb[(size_t)k * C_DIM + t];          // dst
            for (int i = 0; i < cnt; ++i) {
                int u = s_list[off + i];
                acc += xb[(size_t)(KEEP + u) * C_DIM + t];  // src rows, fixed order
            }
            ob[(size_t)k * C_DIM + t] = acc / (float)(cnt + 1);
        }
    }
}

extern "C" void kernel_launch(void* const* d_in, const int* in_sizes, int n_in,
                              void* d_out, int out_size, void* d_ws, size_t ws_size,
                              hipStream_t stream) {
    const float* x = (const float*)d_in[0];
    // d_in[1] = kept_number (== 256, baked into the kernels)
    (void)in_sizes; (void)n_in; (void)out_size; (void)ws_size;

    char*  ws   = (char*)d_ws;
    size_t off0 = 0;                                         // partials: 512*8*192 f32
    size_t off1 = off0 + (size_t)B_DIM * NSPLIT * C_DIM * 4; // invn:     512*192   f32
    size_t off2 = off1 + (size_t)B_DIM * C_DIM * 4;          // dst_idx:  512*512   i32
    float* part = (float*)(ws + off0);
    float* invn = (float*)(ws + off1);
    int*   didx = (int*)  (ws + off2);
    float* out  = (float*)d_out;

    k_partial_sumsq<<<dim3(B_DIM, NSPLIT), C_DIM, 0, stream>>>(x, part);
    k_invnorm<<<(B_DIM * C_DIM) / 256, 256, 0, stream>>>(part, invn);
    k_sim_argmax<<<dim3(4, B_DIM), 256, 0, stream>>>(x, invn, didx);
    k_merge<<<B_DIM, 256, 0, stream>>>(x, didx, out);
}
